// Encoder_21706764714368
// MI455X (gfx1250) — compile-verified, matrix operand order checked by reading, not run on hardware
//
#include <hip/hip_runtime.h>
#include <hip/hip_bf16.h>

#define N_NODES 50000
#define T_STEPS 12
#define E_EDGES 800000
#define DIM     128
#define GDIM    512   // 4 * 128 LSTM gates
#define MBLK    80    // rows per GEMM block (5 row tiles); 50000 = 80 * 625
#define NBLK    625

typedef __attribute__((ext_vector_type(16))) _Float16 v16h;
typedef __attribute__((ext_vector_type(8)))  float    v8f;
typedef int b128_t __attribute__((vector_size(16)));   // 16B payload for async LDS copy

// Optional gfx1250 async global->LDS staging (guarded; sync fallback otherwise)
#if defined(__has_builtin)
#  if __has_builtin(__builtin_amdgcn_global_load_async_to_lds_b128) && \
      __has_builtin(__builtin_amdgcn_s_wait_asynccnt)
#    define HAVE_ASYNC_LDS 1
#  endif
#endif
#ifndef HAVE_ASYNC_LDS
#  define HAVE_ASYNC_LDS 0
#endif

// ---------------------------------------------------------------- utilities

__global__ void zero_kernel(float* p, long n) {
    long i = (long)blockIdx.x * blockDim.x + threadIdx.x;
    long stride = (long)gridDim.x * blockDim.x;
    for (; i < n; i += stride) p[i] = 0.0f;
}

__global__ void deg_kernel(const int* __restrict__ src, const int* __restrict__ dst,
                           float* __restrict__ odeg, float* __restrict__ ideg) {
    int e = blockIdx.x * blockDim.x + threadIdx.x;
    if (e < E_EDGES) {
        atomicAdd(&odeg[src[e]], 1.0f);
        atomicAdd(&ideg[dst[e]], 1.0f);
    }
}

__global__ void norm_kernel(float* __restrict__ a, float* __restrict__ b, int n) {
    int i = blockIdx.x * blockDim.x + threadIdx.x;
    if (i < n) {
        a[i] = rsqrtf(fmaxf(a[i], 1.0f));
        b[i] = rsqrtf(fmaxf(b[i], 1.0f));
    }
}

__global__ void bias_sum_kernel(const float* __restrict__ bih, const float* __restrict__ bhh,
                                float* __restrict__ bsum) {
    int i = blockIdx.x * blockDim.x + threadIdx.x;
    if (i < GDIM) bsum[i] = bih[i] + bhh[i];
}

// ------------------------------------------------- weight fragment packing
// Pack matrix B (K x NC, contraction-major) into exact per-lane WMMA
// f16 B-fragment order: out[((ct*KC+kc)*32 + lane)*16 + h] = B[kb][n]
//   v = h/2, p = h%2
//   kb = kc*32 + (v<4 ? 2v : 16+2(v-4)) + 8*(lane/16) + p
//   n  = ct*16 + lane%16

__global__ void pack_g_kernel(const float* __restrict__ W, _Float16* __restrict__ out,
                              int K, int NC) {
    // B[k][n] = W[k*NC + n]  (W already K-major, as in y = m @ W)
    int idx = blockIdx.x * blockDim.x + threadIdx.x;
    int KC = K >> 5;
    int total = KC * (NC >> 4) * 512;
    if (idx >= total) return;
    int hh   = idx & 15;
    int lane = (idx >> 4) & 31;
    int frag = idx >> 9;
    int kc = frag % KC, ct = frag / KC;
    int v = hh >> 1, p = hh & 1;
    int kb = kc * 32 + (v < 4 ? 2 * v : 16 + 2 * (v - 4)) + 8 * (lane >> 4) + p;
    int n  = ct * 16 + (lane & 15);
    out[idx] = (_Float16)W[(size_t)kb * NC + n];
}

__global__ void pack_lstm_kernel(const float* __restrict__ wih, const float* __restrict__ whh,
                                 _Float16* __restrict__ out) {
    // Fused K=256 (x|h), NC=512.  gates = x @ wih^T + h @ whh^T
    // B[k][n] = k<128 ? wih[n*128+k] : whh[n*128+(k-128)]
    int idx = blockIdx.x * blockDim.x + threadIdx.x;
    const int KC = 8;              // 256/32
    int total = KC * 32 * 512;     // 131072
    if (idx >= total) return;
    int hh   = idx & 15;
    int lane = (idx >> 4) & 31;
    int frag = idx >> 9;
    int kc = frag % KC, ct = frag / KC;
    int v = hh >> 1, p = hh & 1;
    int kb = kc * 32 + (v < 4 ? 2 * v : 16 + 2 * (v - 4)) + 8 * (lane >> 4) + p;
    int n  = ct * 16 + (lane & 15);
    float val = (kb < 128) ? wih[(size_t)n * 128 + kb]
                           : whh[(size_t)n * 128 + (kb - 128)];
    out[idx] = (_Float16)val;
}

// ----------------------------------------------------------- graph scatter
// m[dst[e]] += x[src[e]] * norm_src[src[e]]   (32 lanes per edge, 4 floats/lane)

__global__ void scatter_kernel(const float* __restrict__ x, int ldx,
                               const int* __restrict__ src, const int* __restrict__ dst,
                               const float* __restrict__ nsrc, float* __restrict__ m) {
    long tid = (long)blockIdx.x * blockDim.x + threadIdx.x;
    long e = tid >> 5;
    if (e >= E_EDGES) return;
    int c0 = (int)(tid & 31) * 4;
    int s = src[e], d = dst[e];
    float ns = nsrc[s];
    const float4 v = *(const float4*)(x + (size_t)s * ldx + c0);
    float* o = m + (size_t)d * DIM + c0;
    atomicAdd(o + 0, v.x * ns);
    atomicAdd(o + 1, v.y * ns);
    atomicAdd(o + 2, v.z * ns);
    atomicAdd(o + 3, v.w * ns);
}

// ----------------------------------------------------------- conv GEMM+GELU
// y = gelu((m * norm_dst) @ W + b);  optional column mean accumulated into hs.
// 80-row M-block = 5 row tiles; wave w owns column tile w (8 tiles of 16).
// B fragment loaded once per k-chunk, replayed over 5 A fragments (5x reuse).

__device__ __forceinline__ float gelu_exact(float x) {
    return 0.5f * x * (1.0f + erff(x * 0.70710678118654752f));
}

__global__ __launch_bounds__(256)
void conv_gemm_kernel(const float* __restrict__ m, const float* __restrict__ ndst,
                      const _Float16* __restrict__ Wp, const float* __restrict__ bias,
                      float* __restrict__ yout, float* __restrict__ hs) {
    __shared__ __align__(16) _Float16 As[MBLK][136];   // 80 rows x 128 K, padded
    const int tid = threadIdx.x;
    const int M0 = blockIdx.x * MBLK;

    // Stage A tile (f32 -> f16, scaled by norm_dst) into LDS, coalesced.
    #pragma unroll
    for (int i = 0; i < 5; ++i) {
        int row = (tid >> 4) + 16 * i;
        int c0  = (tid & 15) * 8;
        float sc = ndst[M0 + row];
        const float* srcp = m + (size_t)(M0 + row) * DIM + c0;
        #pragma unroll
        for (int j = 0; j < 8; ++j) As[row][c0 + j] = (_Float16)(srcp[j] * sc);
    }
    __syncthreads();

    const int wave  = tid >> 5;          // = column tile (8 tiles of 16 cols)
    const int lane  = tid & 31;
    const int mrow  = lane & 15;
    const int khalf = (lane >> 4) * 8;

    v8f acc[5] = {};
    #pragma unroll
    for (int kc = 0; kc < 4; ++kc) {
        const _Float16* bp = Wp + ((size_t)(wave * 4 + kc) * 32 + lane) * 16;
        v16h b = *(const v16h*)bp;
        #pragma unroll
        for (int rt = 0; rt < 5; ++rt) {
            v16h a{};
            #pragma unroll
            for (int v = 0; v < 8; ++v) {
                int kb = kc * 32 + (v < 4 ? 2 * v : 16 + 2 * (v - 4)) + khalf;
                a[2 * v]     = As[rt * 16 + mrow][kb];
                a[2 * v + 1] = As[rt * 16 + mrow][kb + 1];
            }
            acc[rt] = __builtin_amdgcn_wmma_f32_16x16x32_f16(false, a, false, b,
                                                             (short)0, acc[rt], false, false);
        }
    }

    const int col = wave * 16 + (lane & 15);
    const float bcol = bias[col];
    float csum = 0.0f;
    #pragma unroll
    for (int rt = 0; rt < 5; ++rt) {
        #pragma unroll
        for (int r = 0; r < 8; ++r) {
            int row = M0 + rt * 16 + r + (lane >> 4) * 8;
            float val = gelu_exact(acc[rt][r] + bcol);
            if (yout) yout[(size_t)row * DIM + col] = val;
            csum += val;
        }
    }
    if (hs) {
        csum += __shfl_xor(csum, 16, 32);   // fold the two half-waves (same col)
        if (lane < 16) atomicAdd(hs + col, csum * (1.0f / N_NODES));
    }
}

// ------------------------------------------------------- fused LSTM step
// gates = [x_t | h_prev] @ Wfused + (b_ih + b_hh); full cell update fused.
// 80-row M-block = 5 row tiles. Wave w owns col tiles {w, w+8, w+16, w+24} so
// its 4 gate accumulators per row tile hold matching i,f,g,o entries per lane.
// h state is kept in f16 (GEMM input precision) -> async-copied into LDS.

__device__ __forceinline__ float sigf(float x) { return 1.0f / (1.0f + expf(-x)); }

__global__ __launch_bounds__(256)
void lstm_step_kernel(const float* __restrict__ xt, int ldx,
                      const _Float16* __restrict__ hprev,
                      const _Float16* __restrict__ Wp, const float* __restrict__ bsum,
                      _Float16* __restrict__ hnext, float* __restrict__ cst,
                      float* __restrict__ htacc) {
    __shared__ __align__(16) _Float16 As[MBLK][264];   // 80 rows x 256 K ([x|h])
    const int tid = threadIdx.x;
    const int M0 = blockIdx.x * MBLK;

    // h-half of K: f16 rows, 16B per lane chunk -> async global->LDS if available
#if HAVE_ASYNC_LDS
    #pragma unroll
    for (int i = 0; i < 5; ++i) {
        int idx = tid + 256 * i;     // 0..1279
        int row = idx >> 4;          // 0..79
        int seg = idx & 15;          // 16B segment of the 256B row
        const _Float16* g = hprev + (size_t)(M0 + row) * DIM + seg * 8;
        _Float16* l = &As[row][128 + seg * 8];
        __builtin_amdgcn_global_load_async_to_lds_b128(
            (__attribute__((address_space(1))) b128_t*)g,
            (__attribute__((address_space(3))) b128_t*)l, 0, 0);
    }
#else
    #pragma unroll
    for (int i = 0; i < 5; ++i) {
        int idx = tid + 256 * i;
        int row = idx >> 4;
        int seg = idx & 15;
        const uint4* g = (const uint4*)(hprev + (size_t)(M0 + row) * DIM + seg * 8);
        *(uint4*)&As[row][128 + seg * 8] = *g;
    }
#endif
    // x-half of K: f32 -> f16 conversion staging
    #pragma unroll
    for (int i = 0; i < 5; ++i) {
        int row = (tid >> 4) + 16 * i;
        int c0  = (tid & 15) * 8;
        const float* sp = xt + (size_t)(M0 + row) * ldx + c0;
        #pragma unroll
        for (int j = 0; j < 8; ++j) As[row][c0 + j] = (_Float16)sp[j];
    }
#if HAVE_ASYNC_LDS
    __builtin_amdgcn_s_wait_asynccnt(0);
#endif
    __syncthreads();

    const int wave  = tid >> 5;
    const int lane  = tid & 31;
    const int mrow  = lane & 15;
    const int khalf = (lane >> 4) * 8;

    v8f acc[20] = {};                      // [rt][gate] = acc[rt*4+q]
    #pragma unroll
    for (int kc = 0; kc < 8; ++kc) {
        v16h b[4];
        #pragma unroll
        for (int q = 0; q < 4; ++q) {
            int ct = wave + 8 * q;
            b[q] = *(const v16h*)(Wp + ((size_t)(ct * 8 + kc) * 32 + lane) * 16);
        }
        #pragma unroll
        for (int rt = 0; rt < 5; ++rt) {
            v16h a{};
            #pragma unroll
            for (int v = 0; v < 8; ++v) {
                int kb = kc * 32 + (v < 4 ? 2 * v : 16 + 2 * (v - 4)) + khalf;
                a[2 * v]     = As[rt * 16 + mrow][kb];
                a[2 * v + 1] = As[rt * 16 + mrow][kb + 1];
            }
            #pragma unroll
            for (int q = 0; q < 4; ++q) {
                acc[rt * 4 + q] = __builtin_amdgcn_wmma_f32_16x16x32_f16(
                    false, a, false, b[q], (short)0, acc[rt * 4 + q], false, false);
            }
        }
    }

    const int col = wave * 16 + (lane & 15);     // gate-local column 0..127
    const float bi = bsum[col];
    const float bf = bsum[128 + col];
    const float bg = bsum[256 + col];
    const float bo = bsum[384 + col];
    #pragma unroll
    for (int rt = 0; rt < 5; ++rt) {
        #pragma unroll
        for (int r = 0; r < 8; ++r) {
            size_t row = (size_t)(M0 + rt * 16 + r + (lane >> 4) * 8);
            size_t idx = row * DIM + col;
            float iv = sigf(acc[rt * 4 + 0][r] + bi);
            float fv = sigf(acc[rt * 4 + 1][r] + bf);
            float gv = tanhf(acc[rt * 4 + 2][r] + bg);
            float ov = sigf(acc[rt * 4 + 3][r] + bo);
            float cn = fv * cst[idx] + iv * gv;
            float hn = ov * tanhf(cn);
            cst[idx]   = cn;
            hnext[idx] = (_Float16)hn;
            htacc[idx] += hn * (1.0f / T_STEPS);
        }
    }
}

// ---------------------------------------------------------------- launcher

extern "C" void kernel_launch(void* const* d_in, const int* in_sizes, int n_in,
                              void* d_out, int out_size, void* d_ws, size_t ws_size,
                              hipStream_t stream) {
    const float* h   = (const float*)d_in[0];
    const int*   src = (const int*)  d_in[1];
    const int*   dst = (const int*)  d_in[2];
    const float* Wg1 = (const float*)d_in[3];
    const float* bg1 = (const float*)d_in[4];
    const float* Wg2 = (const float*)d_in[5];
    const float* bg2 = (const float*)d_in[6];
    const float* wih = (const float*)d_in[7];
    const float* whh = (const float*)d_in[8];
    const float* bih = (const float*)d_in[9];
    const float* bhh = (const float*)d_in[10];
    float* out = (float*)d_out;                    // [hs: 12*128][ht: N*128]
    float* hs_base = out;
    float* ht_acc  = out + T_STEPS * DIM;

    // Workspace carve-out (256B aligned regions)
    char* base = (char*)d_ws;
    size_t off = 0;
    auto carve = [&](size_t bytes) {
        char* p = base + off;
        off = (off + bytes + 255) & ~(size_t)255;
        return p;
    };
    float*    nsrc = (float*)   carve((size_t)N_NODES * 4);
    float*    ndst = (float*)   carve((size_t)N_NODES * 4);
    float*    bufA = (float*)   carve((size_t)N_NODES * DIM * 4);  // m (scatter accum)
    float*    bufB = (float*)   carve((size_t)N_NODES * DIM * 4);  // y1 (conv1 out)
    float*    cbuf = (float*)   carve((size_t)N_NODES * DIM * 4);  // LSTM cell (f32)
    _Float16* hA16 = (_Float16*)carve((size_t)N_NODES * DIM * 2);  // LSTM h ping (f16)
    _Float16* hB16 = (_Float16*)carve((size_t)N_NODES * DIM * 2);  // LSTM h pong (f16)
    _Float16* wg1p = (_Float16*)carve(16384 * 2);
    _Float16* wg2p = (_Float16*)carve(16384 * 2);
    _Float16* wfp  = (_Float16*)carve(131072 * 2);
    float*    bsum = (float*)   carve(GDIM * 4);

    const long ND = (long)N_NODES * DIM;
    const int ZB = 2048;   // grid-stride zero blocks

    // Degrees -> normalization vectors
    zero_kernel<<<ZB, 256, 0, stream>>>(nsrc, N_NODES);
    zero_kernel<<<ZB, 256, 0, stream>>>(ndst, N_NODES);
    deg_kernel<<<E_EDGES / 256, 256, 0, stream>>>(src, dst, nsrc, ndst);
    norm_kernel<<<(N_NODES + 255) / 256, 256, 0, stream>>>(nsrc, ndst, N_NODES);

    // One-time weight packing / bias fuse
    pack_g_kernel<<<16384 / 256, 256, 0, stream>>>(Wg1, wg1p, DIM, DIM);
    pack_g_kernel<<<16384 / 256, 256, 0, stream>>>(Wg2, wg2p, DIM, DIM);
    pack_lstm_kernel<<<131072 / 256, 256, 0, stream>>>(wih, whh, wfp);
    bias_sum_kernel<<<GDIM / 256, 256, 0, stream>>>(bih, bhh, bsum);

    // Zero outputs (hs + ht accumulator live in d_out)
    zero_kernel<<<ZB, 256, 0, stream>>>(out, (long)T_STEPS * DIM + ND);

    const int scatterBlocks = (E_EDGES * 32) / 256;   // 100000

    // ---- Graph phase: 12 timesteps x (conv1 -> gelu -> conv2 -> mean) ----
    for (int t = 0; t < T_STEPS; ++t) {
        // conv1: m = scatter(h[:,t] * nsrc); y1 = gelu((m*ndst) @ Wg1 + bg1)
        zero_kernel<<<ZB, 256, 0, stream>>>(bufA, ND);
        scatter_kernel<<<scatterBlocks, 256, 0, stream>>>(h + (size_t)t * DIM, T_STEPS * DIM,
                                                          src, dst, nsrc, bufA);
        conv_gemm_kernel<<<NBLK, 256, 0, stream>>>(bufA, ndst, wg1p, bg1,
                                                   bufB, (float*)nullptr);
        // conv2: m = scatter(y1 * nsrc); hs[t] = mean(gelu((m*ndst) @ Wg2 + bg2))
        zero_kernel<<<ZB, 256, 0, stream>>>(bufA, ND);
        scatter_kernel<<<scatterBlocks, 256, 0, stream>>>(bufB, DIM, src, dst, nsrc, bufA);
        conv_gemm_kernel<<<NBLK, 256, 0, stream>>>(bufA, ndst, wg2p, bg2,
                                                   (float*)nullptr, hs_base + t * DIM);
    }

    // ---- LSTM phase over raw h; ht = mean_t h_t accumulated into d_out ----
    zero_kernel<<<ZB, 256, 0, stream>>>((float*)hA16, ND / 2);  // h0 = 0 (f16 buffer)
    zero_kernel<<<ZB, 256, 0, stream>>>(cbuf, ND);              // c0 = 0
    for (int t = 0; t < T_STEPS; ++t) {
        _Float16* hp = (t & 1) ? hB16 : hA16;
        _Float16* hn = (t & 1) ? hA16 : hB16;
        lstm_step_kernel<<<NBLK, 256, 0, stream>>>(h + (size_t)t * DIM, T_STEPS * DIM,
                                                   hp, wfp, bsum, hn, cbuf, ht_acc);
    }
}
